// DeepMmdLoss_4320737100084
// MI455X (gfx1250) — compile-verified, matrix operand order checked by reading, not run on hardware
//
#include <hip/hip_runtime.h>
#include <math.h>

// ---------------------------------------------------------------------------
// DeepMMD loss for MI455X (gfx1250, wave32, WMMA).
//
// Sizes: N=4096, D_IN=512, H=10, D_OUT=50. Fea_org = [Xs;Xt] : 8192 x 512.
// Dominant cost: three 4096x4096 Gram matrices over K=512 (org) and K=64
// (deep, padded from 50). Everything is L2-resident (16.8 MB input,
// ~19 MB workspace) -> WMMA-throughput bound, not HBM bound (23.3 TB/s is
// irrelevant here). fp32 accuracy preserved via bf16 hi/lo splitting:
//   x ~= hi + lo,  x.y ~= hi.hi + hi.lo + lo.hi  (3x v_wmma_f32_16x16x32_bf16)
// plus exact fp32 row norms:  D = ||x||^2 + ||y||^2 - 2 x.y
//
// This revision: per-operand lane base pointers hoisted out of the K loops +
// full unroll, so every global_load_b128 in the hot loop uses a compile-time
// immediate offset (no per-iteration v_add_nc_u64 address chains).
// ---------------------------------------------------------------------------

typedef __attribute__((ext_vector_type(16))) __bf16 v16bf;
typedef __attribute__((ext_vector_type(8)))  float  v8f;

#define WMMA_BF16(A, B, C) \
    __builtin_amdgcn_wmma_f32_16x16x32_bf16(false, (A), false, (B), (short)0, (C), false, false)

#define NROWS   8192
#define NHALF   4096
#define DORG    512
#define DDEEP   64      // 50 padded to 64 with zeros
#define DOUT    50
#define HID     10

// ---- workspace layout (bytes) ----
#define OFF_AHI_ORG  0u
#define OFF_ALO_ORG  (OFF_AHI_ORG + (size_t)NROWS * DORG * 2)
#define OFF_AHI_DP   (OFF_ALO_ORG + (size_t)NROWS * DORG * 2)
#define OFF_ALO_DP   (OFF_AHI_DP  + (size_t)NROWS * DDEEP * 2)
#define OFF_NORG     (OFF_ALO_DP  + (size_t)NROWS * DDEEP * 2)
#define OFF_NDP      (OFF_NORG    + (size_t)NROWS * 4)
#define OFF_ACC      (OFF_NDP     + (size_t)NROWS * 4)

__device__ __forceinline__ float softplus_f(float x) {
    // log(1 + exp(x)) == max(x,0) + log1p(exp(-|x|)), overflow safe
    return fmaxf(x, 0.f) + log1pf(expf(-fabsf(x)));
}

__device__ __forceinline__ float wave_sum32(float v) {
    #pragma unroll
    for (int off = 16; off; off >>= 1) v += __shfl_xor(v, off, 32);
    return v;
}

// ---------------------------------------------------------------------------
// Kernel 0: zero the 3 fp64 accumulators (must be re-done every launch).
// ---------------------------------------------------------------------------
__global__ void init_accum_kernel(double* acc) {
    if (threadIdx.x < 3) acc[threadIdx.x] = 0.0;
}

// ---------------------------------------------------------------------------
// Kernel 1: featurize + bf16 hi/lo split + exact fp32 row norms.
// One wave32 per row; 8 waves (256 threads) per block; 1024 blocks.
// ---------------------------------------------------------------------------
__global__ void featurize_kernel(const float* __restrict__ Xs,
                                 const float* __restrict__ Xt,
                                 const float* __restrict__ W1, const float* __restrict__ b1,
                                 const float* __restrict__ W2, const float* __restrict__ b2,
                                 const float* __restrict__ W3, const float* __restrict__ b3,
                                 const float* __restrict__ W4, const float* __restrict__ b4,
                                 __bf16* __restrict__ AhiO, __bf16* __restrict__ AloO,
                                 __bf16* __restrict__ AhiD, __bf16* __restrict__ AloD,
                                 float* __restrict__ nOrg, float* __restrict__ nDp) {
    const int lane = threadIdx.x & 31;
    const int wave = threadIdx.x >> 5;
    const int r    = blockIdx.x * 8 + wave;          // 0..8191
    const float* X = (r < NHALF) ? (Xs + (size_t)r * DORG)
                                 : (Xt + (size_t)(r - NHALF) * DORG);

    // ---- load row (coalesced), norm, bf16 hi/lo split of org features ----
    float xr[16];
    float nrm = 0.f;
    #pragma unroll
    for (int k = 0; k < 16; ++k) {
        float v = X[k * 32 + lane];
        xr[k] = v;
        nrm += v * v;
        __bf16 hi = (__bf16)v;
        float  lo = v - (float)hi;
        size_t idx = (size_t)r * DORG + k * 32 + lane;
        AhiO[idx] = hi;
        AloO[idx] = (__bf16)lo;
    }
    nrm = wave_sum32(nrm);
    if (lane == 0) nOrg[r] = nrm;

    // ---- MLP: 512 -> 10 -> 10 -> 10 -> 50, softplus ----
    float h1[HID], h2[HID], h3[HID];
    #pragma unroll
    for (int j = 0; j < HID; ++j) {
        float a = 0.f;
        #pragma unroll
        for (int k = 0; k < 16; ++k) a += xr[k] * W1[(k * 32 + lane) * HID + j];
        a = wave_sum32(a);                           // all lanes get total
        h1[j] = softplus_f(a + b1[j]);
    }
    #pragma unroll
    for (int j = 0; j < HID; ++j) {
        float a = b2[j];
        #pragma unroll
        for (int i = 0; i < HID; ++i) a += h1[i] * W2[i * HID + j];
        h2[j] = softplus_f(a);
    }
    #pragma unroll
    for (int j = 0; j < HID; ++j) {
        float a = b3[j];
        #pragma unroll
        for (int i = 0; i < HID; ++i) a += h2[i] * W3[i * HID + j];
        h3[j] = softplus_f(a);
    }

    // ---- output layer: lane computes outputs (lane) and (lane+32) ----
    const int o1 = lane, o2 = lane + 32;
    float oa = b4[o1];
    #pragma unroll
    for (int i = 0; i < HID; ++i) oa += h3[i] * W4[i * DOUT + o1];
    float ob = 0.f;
    if (o2 < DOUT) {
        ob = b4[o2];
        #pragma unroll
        for (int i = 0; i < HID; ++i) ob += h3[i] * W4[i * DOUT + o2];
    }
    float nd = wave_sum32(oa * oa + ob * ob);
    if (lane == 0) nDp[r] = nd;

    const size_t db = (size_t)r * DDEEP;
    __bf16 hia = (__bf16)oa;
    AhiD[db + o1] = hia;
    AloD[db + o1] = (__bf16)(oa - (float)hia);
    __bf16 hib = (__bf16)ob;                         // ob==0 for padded cols
    AhiD[db + o2] = hib;
    AloD[db + o2] = (__bf16)(ob - (float)hib);
}

// ---------------------------------------------------------------------------
// WMMA operand layout used here (contiguous k, not AMD's table order): for a Gram matrix the A-operand
// (16x32, M x K) and B-operand (32x16, K x N) per-lane layouts are IDENTICAL:
// lane l<16 holds row l, K kc..kc+15; lane l>=16 holds row l-16, K kc+16..31.
// One 32-byte aligned 32-byte load per lane per fragment.
// Lane base pointer (element offset into a [NROWS x ld] array):
//     (row0 + (lane&15)) * ld + (lane>>4)*16
// Fragment at (row0 + 16*t, kc) is then base + t*16*ld + kc   (imm offset).
// ---------------------------------------------------------------------------
__device__ __forceinline__ const __bf16* frag_base(const __bf16* __restrict__ p,
                                                   int row0, int ld, int lane) {
    return p + (size_t)(row0 + (lane & 15)) * ld + ((lane >> 4) << 4);
}

// ---------------------------------------------------------------------------
// Kernel 2: the Gram/kernel/reduce pass.
// grid = (32, 64, 3); block = 256 threads = 8 waves.
// Block tile: 128 (i) x 64 (j); each wave owns a 32x32 tile (2x2 WMMA tiles).
// z=0: Kx (X,X, weight 2 for i<j), z=1: Ky (Y,Y), z=2: Kxy (full, i!=j).
// ---------------------------------------------------------------------------
__global__ void mmd_pair_kernel(const __bf16* __restrict__ AhiO,
                                const __bf16* __restrict__ AloO,
                                const __bf16* __restrict__ AhiD,
                                const __bf16* __restrict__ AloD,
                                const float* __restrict__ nOrg,
                                const float* __restrict__ nDp,
                                const float* __restrict__ eps_p,
                                const float* __restrict__ sig_p,
                                const float* __restrict__ sig0_p,
                                double* __restrict__ accum) {
    const int z    = blockIdx.z;
    const int lane = threadIdx.x & 31;
    const int wave = threadIdx.x >> 5;
    const int iBlk = blockIdx.x * 128 + (wave & 3) * 32;   // within [0,4096)
    const int jBlk = blockIdx.y * 64  + (wave >> 2) * 32;

    // symmetric matrices: only i<j contributes (weight 2); whole-wave skip
    if (z < 2 && iBlk >= jBlk + 32) return;

    const int offI = (z == 1) ? NHALF : 0;
    const int offJ = (z == 0) ? 0 : NHALF;

    v8f accO[2][2] = {};
    v8f accD[2][2] = {};

    // ---- hoisted lane base pointers (all hot-loop loads use imm offsets) ----
    const __bf16* aOh = frag_base(AhiO, offI + iBlk, DORG, lane);
    const __bf16* aOl = frag_base(AloO, offI + iBlk, DORG, lane);
    const __bf16* bOh = frag_base(AhiO, offJ + jBlk, DORG, lane);
    const __bf16* bOl = frag_base(AloO, offJ + jBlk, DORG, lane);
    const __bf16* aDh = frag_base(AhiD, offI + iBlk, DDEEP, lane);
    const __bf16* aDl = frag_base(AloD, offI + iBlk, DDEEP, lane);
    const __bf16* bDh = frag_base(AhiD, offJ + jBlk, DDEEP, lane);
    const __bf16* bDl = frag_base(AloD, offJ + jBlk, DDEEP, lane);

    // ---- org-feature Gram, K = 512, bf16 hi/lo 3-term split ----
    #pragma unroll
    for (int kc = 0; kc < DORG; kc += 32) {
        v16bf ah[2], al[2], bh[2], bl[2];
        #pragma unroll
        for (int t = 0; t < 2; ++t) {
            ah[t] = *(const v16bf*)(aOh + t * 16 * DORG + kc);
            al[t] = *(const v16bf*)(aOl + t * 16 * DORG + kc);
            bh[t] = *(const v16bf*)(bOh + t * 16 * DORG + kc);
            bl[t] = *(const v16bf*)(bOl + t * 16 * DORG + kc);
        }
        #pragma unroll
        for (int ti = 0; ti < 2; ++ti)
            #pragma unroll
            for (int tj = 0; tj < 2; ++tj) {
                accO[ti][tj] = WMMA_BF16(ah[ti], bh[tj], accO[ti][tj]);
                accO[ti][tj] = WMMA_BF16(ah[ti], bl[tj], accO[ti][tj]);
                accO[ti][tj] = WMMA_BF16(al[ti], bh[tj], accO[ti][tj]);
            }
    }

    // ---- deep-feature Gram, K = 64 (50 zero-padded) ----
    #pragma unroll
    for (int kc = 0; kc < DDEEP; kc += 32) {
        v16bf ah[2], al[2], bh[2], bl[2];
        #pragma unroll
        for (int t = 0; t < 2; ++t) {
            ah[t] = *(const v16bf*)(aDh + t * 16 * DDEEP + kc);
            al[t] = *(const v16bf*)(aDl + t * 16 * DDEEP + kc);
            bh[t] = *(const v16bf*)(bDh + t * 16 * DDEEP + kc);
            bl[t] = *(const v16bf*)(bDl + t * 16 * DDEEP + kc);
        }
        #pragma unroll
        for (int ti = 0; ti < 2; ++ti)
            #pragma unroll
            for (int tj = 0; tj < 2; ++tj) {
                accD[ti][tj] = WMMA_BF16(ah[ti], bh[tj], accD[ti][tj]);
                accD[ti][tj] = WMMA_BF16(ah[ti], bl[tj], accD[ti][tj]);
                accD[ti][tj] = WMMA_BF16(al[ti], bh[tj], accD[ti][tj]);
            }
    }

    // ---- combine: deep kernel + weighted reduction ----
    const float ep     = 1.f / (1.f + expf(-eps_p[0]));
    const float sg     = sig_p[0];
    const float s0     = sig0_p[0];
    const float c_org  = 1.44269504f / (sg * sg);     // exp(-D/sigma)  -> exp2
    const float c_deep = 1.44269504f / (s0 * s0);     // exp(-D/sigma0) -> exp2
    const float one_m_ep = 1.f - ep;

    // C/D layout: lane<16 -> (M = r, N = lane); lane>=16 -> (M = r+8, N = lane-16)
    const int mAdd = ((lane >> 4) << 3);
    const int nCol = (lane & 15);

    float s = 0.f;
    #pragma unroll
    for (int ti = 0; ti < 2; ++ti) {
        #pragma unroll
        for (int tj = 0; tj < 2; ++tj) {
            const int nn  = jBlk + tj * 16 + nCol;
            const float nOj = nOrg[offJ + nn];
            const float nDj = nDp[offJ + nn];
            #pragma unroll
            for (int r = 0; r < 8; ++r) {
                const int m = iBlk + ti * 16 + mAdd + r;
                float Do = fmaxf(nOrg[offI + m] + nOj - 2.f * accO[ti][tj][r], 0.f);
                float Dd = fmaxf(nDp[offI + m]  + nDj - 2.f * accD[ti][tj][r], 0.f);
                float e_org = exp2f(-Do * c_org);
                float val = e_org * (ep + one_m_ep * exp2f(-Dd * c_deep));
                float w = (z < 2) ? ((m < nn) ? 2.f : 0.f)
                                  : ((m != nn) ? 1.f : 0.f);
                s += w * val;
            }
        }
    }
    s = wave_sum32(s);
    if (lane == 0) atomicAdd(&accum[z], (double)s);
}

// ---------------------------------------------------------------------------
// Kernel 3: finalize scalar.
// ---------------------------------------------------------------------------
__global__ void finalize_kernel(const double* __restrict__ accum,
                                float* __restrict__ out) {
    if (threadIdx.x == 0 && blockIdx.x == 0) {
        const double denom = 4096.0 * 4095.0;
        out[0] = (float)((accum[0] + accum[1] - 2.0 * accum[2]) / denom);
    }
}

// ---------------------------------------------------------------------------
extern "C" void kernel_launch(void* const* d_in, const int* in_sizes, int n_in,
                              void* d_out, int out_size, void* d_ws, size_t ws_size,
                              hipStream_t stream) {
    (void)in_sizes; (void)n_in; (void)out_size; (void)ws_size;

    const float* Xs  = (const float*)d_in[0];
    const float* Xt  = (const float*)d_in[1];
    const float* W1  = (const float*)d_in[2];
    const float* b1  = (const float*)d_in[3];
    const float* W2  = (const float*)d_in[4];
    const float* b2  = (const float*)d_in[5];
    const float* W3  = (const float*)d_in[6];
    const float* b3  = (const float*)d_in[7];
    const float* W4  = (const float*)d_in[8];
    const float* b4  = (const float*)d_in[9];
    const float* ep  = (const float*)d_in[10];
    const float* sg  = (const float*)d_in[11];
    const float* sg0 = (const float*)d_in[12];

    char* ws = (char*)d_ws;
    __bf16* AhiO = (__bf16*)(ws + OFF_AHI_ORG);
    __bf16* AloO = (__bf16*)(ws + OFF_ALO_ORG);
    __bf16* AhiD = (__bf16*)(ws + OFF_AHI_DP);
    __bf16* AloD = (__bf16*)(ws + OFF_ALO_DP);
    float*  nOrg = (float*)(ws + OFF_NORG);
    float*  nDp  = (float*)(ws + OFF_NDP);
    double* acc  = (double*)(ws + OFF_ACC);

    init_accum_kernel<<<1, 32, 0, stream>>>(acc);

    featurize_kernel<<<NROWS / 8, 256, 0, stream>>>(
        Xs, Xt, W1, b1, W2, b2, W3, b3, W4, b4,
        AhiO, AloO, AhiD, AloD, nOrg, nDp);

    dim3 grid(NHALF / 128, NHALF / 64, 3);
    mmd_pair_kernel<<<grid, 256, 0, stream>>>(
        AhiO, AloO, AhiD, AloD, nOrg, nDp, ep, sg, sg0, acc);

    finalize_kernel<<<1, 32, 0, stream>>>(acc, (float*)d_out);
}
